// MPBackbone_33560874450991
// MI455X (gfx1250) — matrix operand order repaired and checked by reading, not compiled, not run on hardware
//
#include <hip/hip_runtime.h>
#include <hip/hip_bf16.h>
#include <stdint.h>

typedef __attribute__((ext_vector_type(16))) _Float16 v16h;
typedef __attribute__((ext_vector_type(8)))  float    v8f;

#define NN   10000
#define EE   30000
#define HH   64
#define LL   3
#define KBIG 1088            // 17 * 64
#define KC   34              // KBIG / 32
#define WAVES 8

// ---------------------------------------------------------------------------
// generic helpers
// ---------------------------------------------------------------------------
__global__ __launch_bounds__(256) void k_zero_f32(float* p, int n) {
    int i = blockIdx.x * blockDim.x + threadIdx.x;
    if (i < n) p[i] = 0.0f;
}

__global__ __launch_bounds__(256) void k_degree(const int* __restrict__ ei, float* __restrict__ deg) {
    int e = blockIdx.x * blockDim.x + threadIdx.x;
    if (e < EE) atomicAdd(&deg[ei[EE + e]], 1.0f);
}

__global__ __launch_bounds__(256) void k_invdeg(const float* __restrict__ deg, float* __restrict__ inv) {
    int i = blockIdx.x * blockDim.x + threadIdx.x;
    if (i < NN) { float d = deg[i]; inv[i] = d > 0.0f ? 1.0f / d : 0.0f; }
}

// input MLP: h = relu(x @ w_in + b_in)   [N,16]x[16,64]
__global__ __launch_bounds__(256) void k_input_mlp(const float* __restrict__ x,
                                                   const float* __restrict__ w_in,
                                                   const float* __restrict__ b_in,
                                                   float* __restrict__ h) {
    int idx = blockIdx.x * blockDim.x + threadIdx.x;
    if (idx >= NN * HH) return;
    int node = idx >> 6, o = idx & 63;
    float acc = b_in[o];
#pragma unroll
    for (int j = 0; j < 16; ++j) acc = fmaf(x[node * 16 + j], w_in[j * HH + o], acc);
    h[idx] = fmaxf(acc, 0.0f);
}

// ---------------------------------------------------------------------------
// pre-swizzle Mbig[k=kk*64+i][o] (ew2 with eb2 folded as kk==16) into
// fragment-major f16 layout [l][c][t][lane][16] matching wmma B operand
// ---------------------------------------------------------------------------
__global__ __launch_bounds__(256) void k_swizzle_M(const float* __restrict__ ew2,
                                                   const float* __restrict__ eb2,
                                                   _Float16* __restrict__ Mswz) {
    int gid = blockIdx.x * blockDim.x + threadIdx.x;
    if (gid >= LL * KC * 4 * 32 * 16) return;
    int e    = gid & 15;
    int lane = (gid >> 4) & 31;
    int t    = (gid >> 9) & 3;
    int rest = gid >> 11;          // l*KC + c
    int c    = rest % KC;
    int l    = rest / KC;
    int k    = c * 32 + (lane >> 4) * 16 + e;   // B operand: lanes 0-15 K=0..15, 16-31 K=16..31
    int n    = t * 16 + (lane & 15);
    int kk   = k >> 6, i = k & 63;
    float v  = (kk < 16) ? ew2[((size_t)l * 16 + kk) * 4096 + i * 64 + n]
                         : eb2[(size_t)l * 4096 + i * 64 + n];
    Mswz[gid] = (_Float16)v;
}

__global__ __launch_bounds__(256) void k_swizzle_W(const float* __restrict__ w_root,
                                                   _Float16* __restrict__ Wswz) {
    int gid = blockIdx.x * blockDim.x + threadIdx.x;
    if (gid >= LL * 2 * 4 * 32 * 16) return;
    int e    = gid & 15;
    int lane = (gid >> 4) & 31;
    int t    = (gid >> 9) & 3;
    int rest = gid >> 11;          // l*2 + c
    int c    = rest & 1;
    int l    = rest >> 1;
    int k    = c * 32 + (lane >> 4) * 16 + e;
    int n    = t * 16 + (lane & 15);
    Wswz[gid] = (_Float16)w_root[((size_t)l * 64 + k) * 64 + n];
}

// ---------------------------------------------------------------------------
// fused edge kernel: per 16-edge tile,
//   u_e = [relu(ea@ew1+b1), 1]  (17),  z_e = u_e (x) h[src]  (K=1088)
//   msg[16,64] = z[16,1088] @ Mbig[1088,64]   via v_wmma_f32_16x16x32_f16
//   atomic scatter-add into agg[dst]
// ---------------------------------------------------------------------------
__global__ __launch_bounds__(256) void k_edge_msg(const int* __restrict__ edge_index,
                                                  const float* __restrict__ edge_attr,
                                                  const float* __restrict__ ew1,
                                                  const float* __restrict__ eb1,
                                                  const _Float16* __restrict__ Mswz,
                                                  const float* __restrict__ h,
                                                  float* __restrict__ agg,
                                                  int layer) {
    __shared__ float s_w1[8 * 16];
    __shared__ float s_b1[16];
    __shared__ float s_u[WAVES][16][17];
    __shared__ float s_h[WAVES][16][64];
    __shared__ int   s_dst[WAVES][16];

    const int tid  = threadIdx.x;
    const int wave = tid >> 5;
    const int lane = tid & 31;

    if (tid < 128) s_w1[tid] = ew1[layer * 128 + tid];
    if (tid < 16)  s_b1[tid] = eb1[layer * 16 + tid];
    __syncthreads();

    const int tile   = blockIdx.x * WAVES + wave;
    const bool active = tile < (EE / 16);

    if (active) {
        const int r = lane & 15;
        const int e = tile * 16 + r;
        if (lane < 16) {
            s_dst[wave][r] = edge_index[EE + e];
            float ea[8];
#pragma unroll
            for (int j = 0; j < 8; ++j) ea[j] = edge_attr[e * 8 + j];
#pragma unroll
            for (int o = 0; o < 16; ++o) {
                float acc = s_b1[o];
#pragma unroll
                for (int j = 0; j < 8; ++j) acc = fmaf(ea[j], s_w1[j * 16 + o], acc);
                s_u[wave][r][o] = fmaxf(acc, 0.0f);
            }
            s_u[wave][r][16] = 1.0f;
        }
        // gather h[src]: each half-wave loads 32 floats of its edge row (vectorized)
        const int src = edge_index[tile * 16 + r];
        const int c0  = (lane >> 4) * 32;
        const float4* hp = (const float4*)(h + (size_t)src * HH + c0);
#pragma unroll
        for (int j = 0; j < 8; ++j) ((float4*)&s_h[wave][r][c0])[j] = hp[j];
    }
    __syncthreads();

    if (active) {
        v8f acc[4] = {};
        const int g   = lane >> 4;
        const int row = lane & 15;
        const float* up = s_u[wave][row];
        const float* hp = s_h[wave][row];
        const _Float16* bbase = Mswz + (size_t)layer * KC * 4 * 512;   // 512 halves per (c,t)

        for (int c = 0; c < KC; ++c) {
            if (c + 1 < KC)
                __builtin_prefetch(bbase + ((size_t)(c + 1) * 4) * 512 + lane * 16, 0, 1);
            // build A fragment (outer product u (x) h, on the fly)
            v16h a;
#pragma unroll
            for (int ei = 0; ei < 16; ++ei) {
                int klocal = g * 16 + ei;
                int k = c * 32 + klocal;
                a[ei] = (_Float16)(up[k >> 6] * hp[k & 63]);
            }
#pragma unroll
            for (int t = 0; t < 4; ++t) {
                v16h b = *(const v16h*)(bbase + ((size_t)(c * 4 + t) * 32 + lane) * 16);
                acc[t] = __builtin_amdgcn_wmma_f32_16x16x32_f16(
                    false, a, false, b, (short)0, acc[t], false, false);
            }
        }
        // scatter-add msg into agg[dst]
#pragma unroll
        for (int t = 0; t < 4; ++t) {
            const int n = (lane & 15) + t * 16;
#pragma unroll
            for (int v = 0; v < 8; ++v) {
                const int m = v + 8 * (lane >> 4);
                atomicAdd(&agg[(size_t)s_dst[wave][m] * HH + n], acc[t][v]);
            }
        }
    }
}

// ---------------------------------------------------------------------------
// node update: out = agg*inv_deg + h @ w_root + bias ; accumulate BN stats
// ---------------------------------------------------------------------------
__global__ __launch_bounds__(256) void k_node_update(const float* __restrict__ h,
                                                     const float* __restrict__ agg,
                                                     const float* __restrict__ inv_deg,
                                                     const _Float16* __restrict__ Wswz,
                                                     const float* __restrict__ conv_bias,
                                                     float* __restrict__ out,
                                                     float* __restrict__ bn,   // [0]=sum [64]=sumsq
                                                     int layer) {
    __shared__ float s_h[WAVES][16][64];

    const int tid  = threadIdx.x;
    const int wave = tid >> 5;
    const int lane = tid & 31;
    const int tile = blockIdx.x * WAVES + wave;
    const int base = tile * 16;
    const bool active = base < NN;

    if (active) {
        const int r  = lane & 15;
        const int c0 = (lane >> 4) * 32;
        const float4* hp = (const float4*)(h + (size_t)(base + r) * HH + c0);
#pragma unroll
        for (int j = 0; j < 8; ++j) ((float4*)&s_h[wave][r][c0])[j] = hp[j];
    }
    __syncthreads();

    if (active) {
        v8f acc[4] = {};
        const int g   = lane >> 4;
        const int row = lane & 15;
        const _Float16* wb = Wswz + (size_t)layer * 2 * 4 * 512;
#pragma unroll
        for (int c = 0; c < 2; ++c) {
            v16h a;
#pragma unroll
            for (int ei = 0; ei < 16; ++ei) {
                int klocal = g * 16 + ei;
                a[ei] = (_Float16)s_h[wave][row][c * 32 + klocal];
            }
#pragma unroll
            for (int t = 0; t < 4; ++t) {
                v16h b = *(const v16h*)(wb + ((size_t)(c * 4 + t) * 32 + lane) * 16);
                acc[t] = __builtin_amdgcn_wmma_f32_16x16x32_f16(
                    false, a, false, b, (short)0, acc[t], false, false);
            }
        }
#pragma unroll
        for (int t = 0; t < 4; ++t) {
            const int n = (lane & 15) + t * 16;
            const float bias = conv_bias[layer * HH + n];
            float csum = 0.0f, csq = 0.0f;
#pragma unroll
            for (int v = 0; v < 8; ++v) {
                const int m    = v + 8 * (lane >> 4);
                const int node = base + m;
                float val = acc[t][v] + agg[(size_t)node * HH + n] * inv_deg[node] + bias;
                out[(size_t)node * HH + n] = val;
                csum += val;
                csq  = fmaf(val, val, csq);
            }
            atomicAdd(&bn[n],      csum);
            atomicAdd(&bn[64 + n], csq);
        }
    }
}

__global__ void k_bn_stats(float* bn) {
    int i = threadIdx.x;
    if (i < HH) {
        float mu  = bn[i] * (1.0f / NN);
        float var = bn[64 + i] * (1.0f / NN) - mu * mu;
        bn[128 + i] = mu;
        bn[192 + i] = rsqrtf(var + 1e-5f);
    }
}

__global__ __launch_bounds__(256) void k_finalize(const float* __restrict__ out,
                                                  const float* __restrict__ h,
                                                  const float* __restrict__ bn,
                                                  const float* __restrict__ gamma,
                                                  const float* __restrict__ beta,
                                                  float* __restrict__ dst, int layer) {
    int idx = blockIdx.x * blockDim.x + threadIdx.x;
    if (idx >= NN * HH) return;
    int ch = idx & 63;
    float v = gamma[layer * HH + ch] * (out[idx] - bn[128 + ch]) * bn[192 + ch]
            + beta[layer * HH + ch];
    dst[idx] = fmaxf(v, 0.0f) + h[idx];
}

// ---------------------------------------------------------------------------
// host launch
// ---------------------------------------------------------------------------
extern "C" void kernel_launch(void* const* d_in, const int* in_sizes, int n_in,
                              void* d_out, int out_size, void* d_ws, size_t ws_size,
                              hipStream_t stream) {
    const float* x       = (const float*)d_in[0];
    const int*   ei      = (const int*)  d_in[1];
    const float* ea      = (const float*)d_in[2];
    const float* w_in    = (const float*)d_in[3];
    const float* b_in    = (const float*)d_in[4];
    const float* ew1     = (const float*)d_in[5];
    const float* eb1     = (const float*)d_in[6];
    const float* ew2     = (const float*)d_in[7];
    const float* eb2     = (const float*)d_in[8];
    const float* w_root  = (const float*)d_in[9];
    const float* cbias   = (const float*)d_in[10];
    const float* bn_g    = (const float*)d_in[11];
    const float* bn_b    = (const float*)d_in[12];

    char* ws = (char*)d_ws;
    const size_t NH = (size_t)NN * HH * sizeof(float);     // 2,560,000
    float*     h_buf  = (float*)(ws);
    float*     outb   = (float*)(ws + NH);
    float*     aggb   = (float*)(ws + 2 * NH);
    float*     degb   = (float*)(ws + 3 * NH);                       // NN floats
    float*     invd   = (float*)(ws + 3 * NH + 40192);               // NN floats
    float*     bnb    = (float*)(ws + 3 * NH + 2 * 40192);           // 256 floats
    _Float16*  Mswz   = (_Float16*)(ws + 3 * NH + 2 * 40192 + 1024);
    _Float16*  Wswz   = (_Float16*)(ws + 3 * NH + 2 * 40192 + 1024 + (size_t)LL * KC * 4 * 512 * 2);

    const int T = 256;
    // degree / inv-degree
    k_zero_f32<<<(NN + T - 1) / T, T, 0, stream>>>(degb, NN);
    k_degree<<<(EE + T - 1) / T, T, 0, stream>>>(ei, degb);
    k_invdeg<<<(NN + T - 1) / T, T, 0, stream>>>(degb, invd);
    // input MLP
    k_input_mlp<<<(NN * HH + T - 1) / T, T, 0, stream>>>(x, w_in, b_in, h_buf);
    // weight pre-swizzle (fragment-major f16)
    k_swizzle_M<<<(LL * KC * 4 * 32 * 16 + T - 1) / T, T, 0, stream>>>(ew2, eb2, Mswz);
    k_swizzle_W<<<(LL * 2 * 4 * 32 * 16 + T - 1) / T, T, 0, stream>>>(w_root, Wswz);

    const int edgeBlocks = ((EE / 16) + WAVES - 1) / WAVES;          // 235
    const int nodeBlocks = ((NN + 15) / 16 + WAVES - 1) / WAVES;     // 79

    for (int l = 0; l < LL; ++l) {
        k_zero_f32<<<(NN * HH + T - 1) / T, T, 0, stream>>>(aggb, NN * HH);
        k_zero_f32<<<1, T, 0, stream>>>(bnb, 128);
        k_edge_msg<<<edgeBlocks, T, 0, stream>>>(ei, ea, ew1, eb1, Mswz, h_buf, aggb, l);
        k_node_update<<<nodeBlocks, T, 0, stream>>>(h_buf, aggb, invd, Wswz, cbias, outb, bnb, l);
        k_bn_stats<<<1, 64, 0, stream>>>(bnb);
        float* dst = (l == LL - 1) ? (float*)d_out : h_buf;
        k_finalize<<<(NN * HH + T - 1) / T, T, 0, stream>>>(outb, h_buf, bnb, bn_g, bn_b, dst, l);
    }
}
